// BasicBlockLogS_37709812859186
// MI455X (gfx1250) — matrix operand order repaired and checked by reading, not compiled, not run on hardware
//
#include <hip/hip_runtime.h>
#include <hip/hip_bf16.h>

// ---------------------------------------------------------------------------
// BasicBlockLogS fused pipeline for gfx1250 (MI455X):
//   log-polar pooling  ->  bf16 WMMA implicit GEMM (K = 256 ch * 13 bins)
//   -> deterministic BN stats -> fused BN+residual+ReLU.
// ---------------------------------------------------------------------------

typedef __attribute__((ext_vector_type(16))) __bf16 v16bf;
typedef __attribute__((ext_vector_type(8)))  __bf16 v8bf;
typedef __attribute__((ext_vector_type(8)))  float  v8f;

#define C_      256
#define H_      28
#define W_      28
#define B_      32
#define PIX     (H_ * W_)          // 784
#define NPIX    (B_ * PIX)         // 25088
#define NBIN    13                 // 12 log-polar bins + center
#define KTOT    (C_ * NBIN)        // 3328
#define KSTEPS  (KTOT / 32)        // 104
#define CHUNKB  8                  // images per pooling/GEMM chunk
#define NCHUNK  (B_ / CHUNKB)      // 4
#define CHUNKM  (CHUNKB * PIX)     // 6272 pixels per chunk (49 * 128)
#define LDA     48                 // padded LDS row stride (elements), 96 B
#define NSLAB   (NPIX / 256)       // 98 stat slabs

// ---------------------------------------------------------------------------
// 1) Build log-polar bin tables with float64 math (matches numpy reference).
// ---------------------------------------------------------------------------
__global__ void build_tables(float* __restrict__ norm,
                             int* __restrict__ binStart,
                             int* __restrict__ tapRel) {
  if (threadIdx.x != 0 || blockIdx.x != 0) return;
  int bins[81], rel[81], cnt[12];
  for (int i = 0; i < 12; ++i) cnt[i] = 0;
  int n = 0;
  const double twopi = 6.283185307179586476925286766559;
  for (int i = 0; i < 9; ++i) {
    for (int j = 0; j < 9; ++j) {
      const int dy = i - 4, dx = j - 4;
      const double r = sqrt((double)(dy * dy + dx * dx));
      if (r < 1.0) continue;                       // center handled separately
      int lev = (int)floor(log(r) / log(2.0));
      lev = lev < 0 ? 0 : (lev > 1 ? 1 : lev);
      double th = atan2((double)dy, (double)dx);
      if (th < 0.0) th += twopi;
      int ang = (int)(th / (twopi / 6.0));
      if (ang > 5) ang = 5;
      const int b = lev * 6 + ang;
      bins[n] = b;
      rel[n]  = dy * 36 + dx;                      // offset in 36-wide padded tile
      cnt[b]++;
      n++;
    }
  }
  int start = 0;
  int pos[12];
  for (int b = 0; b < 12; ++b) { binStart[b] = start; pos[b] = start; start += cnt[b]; }
  binStart[12] = start;                            // == 80
  for (int t = 0; t < n; ++t) tapRel[pos[bins[t]]++] = rel[t];
  for (int b = 0; b < 12; ++b) norm[b] = 1.0f / (float)(cnt[b] < 1 ? 1 : cnt[b]);
}

// ---------------------------------------------------------------------------
// 2) Pack weights: Wb[n][k], k = c*13 + bin  (bf16, mask-normalization folded).
// ---------------------------------------------------------------------------
__global__ __launch_bounds__(256) void pack_weights(const float* __restrict__ wconv1,
                                                    const float* __restrict__ wcenter,
                                                    const float* __restrict__ norm,
                                                    __bf16* __restrict__ wb) {
  const int n = blockIdx.x;      // output channel p
  const int c = threadIdx.x;     // input channel
  const size_t base  = (size_t)n * KTOT + (size_t)c * NBIN;
  const size_t wbase = ((size_t)n * C_ + c) * 12;
  for (int b = 0; b < 12; ++b)
    wb[base + b] = (__bf16)(wconv1[wbase + b] * norm[b]);
  wb[base + 12] = (__bf16)wcenter[(size_t)n * C_ + c];
}

// ---------------------------------------------------------------------------
// 3) Pooling: per (image, 8-channel group) block.  x planes staged in LDS,
//    13 bin sums per (pixel, channel) written as bf16 into P13[pixel][k].
// ---------------------------------------------------------------------------
#define PLANE  1296                 // 36*36
#define PLANEP 1300                 // padded to dodge LDS bank aliasing
__global__ __launch_bounds__(256) void pool_kernel(const float* __restrict__ x,
                                                   const int* __restrict__ binStart_g,
                                                   const int* __restrict__ tapRel_g,
                                                   __bf16* __restrict__ p13,
                                                   int chunk) {
  __shared__ float tile[8 * PLANEP];
  __shared__ int   sStart[13];
  __shared__ int   sRel[80];
  const int t      = threadIdx.x;
  const int bLocal = blockIdx.x >> 5;        // 0..7 image within chunk
  const int cg     = blockIdx.x & 31;        // 0..31 channel group (8 channels)
  const int b      = chunk * CHUNKB + bLocal;
  const int c0     = cg * 8;

  if (t < 13) sStart[t] = binStart_g[t];
  if (t < 80) sRel[t]   = tapRel_g[t];

  // load 8 zero-padded 36x36 planes
  for (int i = t; i < 8 * PLANE; i += 256) {
    const int pl = i / PLANE, r = i - pl * PLANE;
    const int py = r / 36, px = r - py * 36;
    const int gy = py - 4, gx = px - 4;
    float v = 0.f;
    if (gy >= 0 && gy < H_ && gx >= 0 && gx < W_)
      v = x[(((size_t)b * C_ + (c0 + pl)) * PIX) + gy * W_ + gx];
    tile[pl * PLANEP + r] = v;
  }
  __syncthreads();

  // 784 pixels * 8 channels = 6272 tasks; channel index fastest so that the
  // 26 B stores of 8 neighboring lanes form 208 B contiguous runs.
  for (int idx = t; idx < PIX * 8; idx += 256) {
    const int c8  = idx & 7;
    const int pix = idx >> 3;
    const int h   = pix / W_, wc = pix - h * W_;
    const float* tp = &tile[c8 * PLANEP + (h + 4) * 36 + (wc + 4)];
    const size_t obase = ((size_t)(bLocal * PIX + pix)) * KTOT + (size_t)(c0 + c8) * NBIN;
    for (int bn = 0; bn < 12; ++bn) {
      float s = 0.f;
      const int e = sStart[bn + 1];
      for (int i = sStart[bn]; i < e; ++i) s += tp[sRel[i]];
      p13[obase + bn] = (__bf16)s;
    }
    p13[obase + 12] = (__bf16)tp[0];           // center tap
  }
}

// ---------------------------------------------------------------------------
// 4) GEMM: out1[pixel][p] = P13[pixel][:] * Wb[p][:]^T + b_center[p]
//    BM=128, BN=64, 256 threads = 8 waves (4 M-waves x 2 N-waves),
//    wave tile 32x32 = 2x2 WMMA f32_16x16x32_bf16 accumulators.
// ---------------------------------------------------------------------------
__global__ __launch_bounds__(256) void gemm_kernel(const __bf16* __restrict__ p13,
                                                   const __bf16* __restrict__ wb,
                                                   const float* __restrict__ bias,
                                                   float* __restrict__ out1,
                                                   int chunk) {
  __shared__ __bf16 As[2][128 * LDA];
  __shared__ __bf16 Bs[2][64 * LDA];
  const int t    = threadIdx.x;
  const int lane = t & 31;
  const int wave = t >> 5;
  const int wm   = wave & 3;              // M sub-tile (x32)
  const int wn   = wave >> 2;             // N sub-tile (x32)
  const int half = lane >> 4;             // lane half for WMMA K interleave
  const int l16  = lane & 15;
  const int mTile = blockIdx.x * 128;     // chunk-local pixel base
  const int nTile = blockIdx.y * 64;      // output-channel base

  v8f acc[2][2];
  for (int i = 0; i < 2; ++i)
    for (int j = 0; j < 2; ++j)
      for (int e = 0; e < 8; ++e) acc[i][j][e] = 0.f;

  auto loadTile = [&](int ks, int buf) {
    const int kBase = ks * 32;
    // A: 128 rows x 64 B  (two 16 B segments per thread)
#pragma unroll
    for (int p = 0; p < 2; ++p) {
      const int idx = t + p * 256;
      const int row = idx >> 2, seg = idx & 3;
      const uint4 v = *reinterpret_cast<const uint4*>(
          p13 + (size_t)(mTile + row) * KTOT + kBase + seg * 8);
      *reinterpret_cast<uint4*>(&As[buf][row * LDA + seg * 8]) = v;
    }
    // B: 64 rows x 64 B (one 16 B segment per thread)
    {
      const int row = t >> 2, seg = t & 3;
      const uint4 v = *reinterpret_cast<const uint4*>(
          wb + (size_t)(nTile + row) * KTOT + kBase + seg * 8);
      *reinterpret_cast<uint4*>(&Bs[buf][row * LDA + seg * 8]) = v;
    }
  };

  loadTile(0, 0);
  __syncthreads();

  for (int ks = 0; ks < KSTEPS; ++ks) {
    const int buf = ks & 1;
    if (ks + 1 < KSTEPS) loadTile(ks + 1, buf ^ 1);
    if (ks + 2 < KSTEPS)  // keep the A stream warm two K-steps ahead
      __builtin_prefetch(p13 + (size_t)(mTile + (t >> 2)) * KTOT + (ks + 2) * 32, 0, 0);

    const __bf16* Ab = &As[buf][0];
    const __bf16* Bb = &Bs[buf][0];
    // A fragment: ISA 16-bit A layout -> elems 0-7 at K=half*8, 8-15 at K=16+half*8
    v16bf a0, a1;
    {
      const int r0 = (wm * 32 + l16) * LDA + half * 8;
      ((v8bf*)&a0)[0] = *(const v8bf*)(Ab + r0);
      ((v8bf*)&a0)[1] = *(const v8bf*)(Ab + r0 + 16);
      const int r1 = r0 + 16 * LDA;
      ((v8bf*)&a1)[0] = *(const v8bf*)(Ab + r1);
      ((v8bf*)&a1)[1] = *(const v8bf*)(Ab + r1 + 16);
    }
    // B fragment: same K order as A -> elems 0-7 at K=half*8, 8-15 at K=16+half*8
    const int b0r = (wn * 32 + l16) * LDA + half * 8;
    auto ldB = [&](const __bf16* p) { v16bf v; ((v8bf*)&v)[0] = *(const v8bf*)(p); ((v8bf*)&v)[1] = *(const v8bf*)(p + 16); return v; };
    const v16bf b0 = ldB(Bb + b0r);
    const v16bf b1 = ldB(Bb + b0r + 16 * LDA);

    acc[0][0] = __builtin_amdgcn_wmma_f32_16x16x32_bf16(false, a0, false, b0, (short)0, acc[0][0], false, false);
    acc[0][1] = __builtin_amdgcn_wmma_f32_16x16x32_bf16(false, a0, false, b1, (short)0, acc[0][1], false, false);
    acc[1][0] = __builtin_amdgcn_wmma_f32_16x16x32_bf16(false, a1, false, b0, (short)0, acc[1][0], false, false);
    acc[1][1] = __builtin_amdgcn_wmma_f32_16x16x32_bf16(false, a1, false, b1, (short)0, acc[1][1], false, false);
    __syncthreads();
  }

  // epilogue: D elem r -> (M = r + half*8, N = l16); coalesced f32 stores
#pragma unroll
  for (int i = 0; i < 2; ++i) {
    const size_t pixBase = (size_t)chunk * CHUNKM + mTile + wm * 32 + i * 16 + half * 8;
#pragma unroll
    for (int j = 0; j < 2; ++j) {
      const int nCol = nTile + wn * 32 + j * 16 + l16;
      const float bv = bias[nCol];
#pragma unroll
      for (int r = 0; r < 8; ++r)
        out1[(pixBase + r) * C_ + nCol] = acc[i][j][r] + bv;
    }
  }
}

// ---------------------------------------------------------------------------
// 5) BN stats, two-stage deterministic reduction (no float atomics).
// ---------------------------------------------------------------------------
__global__ __launch_bounds__(256) void stats_partial(const float* __restrict__ out1,
                                                     float* __restrict__ partial) {
  const int ch = threadIdx.x;
  float s = 0.f, ss = 0.f;
  const size_t base = (size_t)blockIdx.x * 256;
  for (int i = 0; i < 256; ++i) {
    const float v = out1[(base + i) * C_ + ch];     // fully coalesced
    s += v;
    ss += v * v;
  }
  partial[(size_t)blockIdx.x * 512 + ch]       = s;
  partial[(size_t)blockIdx.x * 512 + 256 + ch] = ss;
}

__global__ __launch_bounds__(256) void bn_reduce(const float* __restrict__ partial,
                                                 const float* __restrict__ gamma,
                                                 const float* __restrict__ beta,
                                                 float* __restrict__ scale,
                                                 float* __restrict__ shift) {
  const int p = threadIdx.x;
  float s = 0.f, ss = 0.f;
  for (int k = 0; k < NSLAB; ++k) {
    s  += partial[(size_t)k * 512 + p];
    ss += partial[(size_t)k * 512 + 256 + p];
  }
  const float invN = 1.0f / (float)NPIX;
  const float mean = s * invN;
  const float var  = ss * invN - mean * mean;
  const float inv  = rsqrtf(var + 1e-5f);
  const float a    = gamma[p] * inv;
  scale[p] = a;
  shift[p] = beta[p] - mean * a;
}

// ---------------------------------------------------------------------------
// 6) Finalize: BN affine + residual + ReLU, LDS transpose so both the
//    [pixel][p] reads and the NCHW writes are coalesced.
// ---------------------------------------------------------------------------
__global__ __launch_bounds__(256) void finalize_kernel(const float* __restrict__ out1,
                                                       const float* __restrict__ x,
                                                       const float* __restrict__ scale,
                                                       const float* __restrict__ shift,
                                                       float* __restrict__ out) {
  __shared__ float tile[16][257];        // +1 pad: pixel stride = 1 bank
  const int g = blockIdx.x;              // 16-pixel group within image (49)
  const int b = blockIdx.y;              // image
  const int t = threadIdx.x;
  for (int k = 0; k < 16; ++k)
    tile[k][t] = out1[((size_t)b * PIX + g * 16 + k) * C_ + t];
  __syncthreads();
  const int pixLane = t & 15;
  const int chRow   = t >> 4;
  const size_t pixIdx = (size_t)g * 16 + pixLane;
#pragma unroll
  for (int pc = 0; pc < 16; ++pc) {
    const int p = pc * 16 + chRow;
    const size_t o = ((size_t)b * C_ + p) * PIX + pixIdx;
    float v = tile[pixLane][p];
    v = v * scale[p] + shift[p] + x[o];
    out[o] = fmaxf(v, 0.f);
  }
}

// ---------------------------------------------------------------------------
// Host: workspace carve-up + launch sequence (all on `stream`).
// ---------------------------------------------------------------------------
extern "C" void kernel_launch(void* const* d_in, const int* in_sizes, int n_in,
                              void* d_out, int out_size, void* d_ws, size_t ws_size,
                              hipStream_t stream) {
  (void)in_sizes; (void)n_in; (void)out_size; (void)ws_size;
  const float* x        = (const float*)d_in[0];
  const float* w_conv1  = (const float*)d_in[1];
  const float* w_center = (const float*)d_in[2];
  const float* b_center = (const float*)d_in[3];
  const float* gamma    = (const float*)d_in[4];
  const float* beta     = (const float*)d_in[5];
  float* out = (float*)d_out;

  char* ws = (char*)d_ws;
  size_t off = 0;
  auto take = [&](size_t bytes) {
    char* p = ws + off;
    off = (off + bytes + 255) & ~(size_t)255;
    return p;
  };
  float*  norm     = (float*)take(12 * sizeof(float));
  int*    binStart = (int*)take(13 * sizeof(int));
  int*    tapRel   = (int*)take(80 * sizeof(int));
  float*  partial  = (float*)take((size_t)NSLAB * 512 * sizeof(float));
  float*  scale    = (float*)take(C_ * sizeof(float));
  float*  shift    = (float*)take(C_ * sizeof(float));
  __bf16* Wb       = (__bf16*)take((size_t)C_ * KTOT * sizeof(__bf16));
  float*  out1     = (float*)take((size_t)NPIX * C_ * sizeof(float));
  __bf16* P13      = (__bf16*)take((size_t)CHUNKM * KTOT * sizeof(__bf16));

  build_tables<<<1, 1, 0, stream>>>(norm, binStart, tapRel);
  pack_weights<<<C_, 256, 0, stream>>>(w_conv1, w_center, norm, Wb);

  for (int chunk = 0; chunk < NCHUNK; ++chunk) {
    pool_kernel<<<CHUNKB * 32, 256, 0, stream>>>(x, binStart, tapRel, P13, chunk);
    gemm_kernel<<<dim3(CHUNKM / 128, C_ / 64), 256, 0, stream>>>(P13, Wb, b_center, out1, chunk);
  }

  stats_partial<<<NSLAB, 256, 0, stream>>>(out1, partial);
  bn_reduce<<<1, 256, 0, stream>>>(partial, gamma, beta, scale, shift);
  finalize_kernel<<<dim3(PIX / 16, B_), 256, 0, stream>>>(out1, x, scale, shift, out);
}
